// NeuralRoughSimulator_23072564314253
// MI455X (gfx1250) — compile-verified, matrix operand order checked by reading, not run on hardware
//
#include <hip/hip_runtime.h>
#include <hip/hip_bf16.h>

typedef __attribute__((ext_vector_type(16))) _Float16 v16h;
typedef __attribute__((ext_vector_type(8)))  _Float16 v8h;
typedef __attribute__((ext_vector_type(8)))  float    v8f;

#define BATCH        32768
#define NSTEPS       512
#define ROWS_PER_WG  128
#define WAVES        4
#define HSTR         72   // padded row stride (halfs) for activations/weights -> bank-conflict-free b128 frags
#define CSTR         72

struct Params {
    const float* W[2][4];   // [mlp][layer] weights, row-major [out,in]
    const float* Bb[2][4];  // [mlp][layer] biases
};

struct __align__(16) Smem {
    _Float16 Wh[2][2][64 * HSTR];            // hidden 64x64 weights, f16, padded rows
    _Float16 c1[2][ROWS_PER_WG * CSTR];      // per-path constant layer-0 preactivation
    _Float16 h[WAVES][32 * HSTR];            // per-wave activation tile (32 rows x 64)
    _Float16 wlv[2][64];                     // W0[:,6] (log_v column)
    _Float16 w4[2][64];                      // head weights
    float    bh[2][2][64];                   // hidden biases (f32)
    float    b4[2];                          // head bias
};

__device__ __forceinline__ float fast_rcp(float x) { return __builtin_amdgcn_rcpf(x); }

// Native V_TANH_F32 on CDNA5 TRANS unit when the builtin exists; guarded fallback otherwise.
__device__ __forceinline__ float fast_tanh(float x) {
#if __has_builtin(__builtin_amdgcn_tanhf)
    return __builtin_amdgcn_tanhf(x);
#elif __has_builtin(__builtin_amdgcn_tanh_f32)
    return __builtin_amdgcn_tanh_f32(x);
#else
    float ax = __builtin_fabsf(x);
    float e  = __expf(2.0f * ax);                 // v_exp_f32 path
    float t  = 1.0f - 2.0f * fast_rcp(e + 1.0f);  // saturates cleanly to 1 for large |x|
    return __builtin_copysignf(t, x);
#endif
}
__device__ __forceinline__ float fast_sigmoid(float x) {
    return fast_rcp(1.0f + __expf(-x));
}

// Load a 16-lane-row fragment: K packed as [kbase+8h .. +7] in elems 0..7,
// [kbase+16+8h .. +7] in elems 8..15 (ISA 7.12.2, 16-bit A 16x32; B mirrored over W rows).
__device__ __forceinline__ v16h load_frag(const _Float16* mat, int row, int kbase, int h8) {
    const _Float16* p = mat + row * HSTR;
    union { v16h v; v8h q[2]; } u;
    u.q[0] = *(const v8h*)(p + kbase + h8);
    u.q[1] = *(const v8h*)(p + kbase + 16 + h8);
    return u.v;
}

// One 64x64 hidden layer over the wave's 32-row tile: D = tanh(h @ W^T + b), in place.
__device__ __forceinline__ void hidden_layer(_Float16* hb, const _Float16* Wl,
                                             const float* bl, int lane) {
    const int nrow = lane & 15;
    const int h8   = (lane >> 4) << 3;
    const v8f vz = {0.f,0.f,0.f,0.f,0.f,0.f,0.f,0.f};
    v8f acc[2][4];
#pragma unroll
    for (int mb = 0; mb < 2; ++mb)
#pragma unroll
        for (int nt = 0; nt < 4; ++nt) acc[mb][nt] = vz;

#pragma unroll
    for (int kc = 0; kc < 2; ++kc) {
        const int kb = kc * 32;
        v16h a0 = load_frag(hb, nrow,      kb, h8);
        v16h a1 = load_frag(hb, 16 + nrow, kb, h8);
#pragma unroll
        for (int nt = 0; nt < 4; ++nt) {
            v16h bf = load_frag(Wl, nt * 16 + nrow, kb, h8);
            acc[0][nt] = __builtin_amdgcn_wmma_f32_16x16x32_f16(
                false, a0, false, bf, (short)0, acc[0][nt], false, false);
            acc[1][nt] = __builtin_amdgcn_wmma_f32_16x16x32_f16(
                false, a1, false, bf, (short)0, acc[1][nt], false, false);
        }
    }
    // Epilogue: D layout = (N = lane%16, M = r + 8*(lane/16)); bias + tanh, store f16.
#pragma unroll
    for (int mb = 0; mb < 2; ++mb)
#pragma unroll
        for (int nt = 0; nt < 4; ++nt) {
            const int n = nt * 16 + nrow;
            const float bias = bl[n];
            union { v8f v; float f[8]; } u; u.v = acc[mb][nt];
            _Float16* cp = hb + (mb * 16 + h8) * HSTR + n;
#pragma unroll
            for (int r = 0; r < 8; ++r)
                cp[r * HSTR] = (_Float16)fast_tanh(u.f[r] + bias);
        }
}

__global__ __launch_bounds__(128)
void nrs_kernel(const float* __restrict__ init_var, const float* __restrict__ sig,
                const float* __restrict__ bi, const float* __restrict__ dtp,
                Params prm, float* __restrict__ out) {
    __shared__ Smem sm;
    const int tid  = threadIdx.x;
    const int lane = tid & 31;
    const int w    = tid >> 5;
    const int grow = blockIdx.x * ROWS_PER_WG + tid;

    // ---- one-time staging: weights -> LDS (f16), c1 precompute ----
#pragma unroll
    for (int p = 0; p < 2; ++p) {
#pragma unroll
        for (int l = 0; l < 2; ++l) {
            const float* Ws = prm.W[p][l + 1];
            for (int i = tid; i < 4096; i += 128) {
                int r = i >> 6, c = i & 63;
                sm.Wh[p][l][r * HSTR + c] = (_Float16)Ws[i];
            }
            const float* bs = prm.Bb[p][l + 1];
            for (int i = tid; i < 64; i += 128) sm.bh[p][l][i] = bs[i];
        }
        const float* W0 = prm.W[p][0];
        const float* W3 = prm.W[p][3];
        for (int i = tid; i < 64; i += 128) {
            sm.wlv[p][i] = (_Float16)W0[i * 7 + 6];
            sm.w4[p][i]  = (_Float16)W3[i];
        }
        if (tid == 0) sm.b4[p] = prm.Bb[p][3][0];
    }
    {
        float sg[6];
#pragma unroll
        for (int s = 0; s < 6; ++s) sg[s] = sig[(size_t)grow * 6 + s];
#pragma unroll
        for (int p = 0; p < 2; ++p) {
            const float* W0 = prm.W[p][0];
            const float* b0 = prm.Bb[p][0];
            for (int j = 0; j < 64; ++j) {
                float a = b0[j];
#pragma unroll
                for (int s = 0; s < 6; ++s) a += W0[j * 7 + s] * sg[s];
                sm.c1[p][tid * CSTR + j] = (_Float16)a;
            }
        }
    }
    __syncthreads();

    const float dt = dtp[0];
    float sv = init_var[grow];
    sv = fminf(fmaxf(sv, 0.005f), 0.2f);
    float lv = __logf(sv);

    _Float16* hb    = sm.h[w];
    _Float16* myrow = hb + lane * HSTR;

    for (int step = 0; step < NSTEPS; ++step) {
        const float dw = __builtin_nontemporal_load(bi + (size_t)grow * NSTEPS + step);
        float outm[2];
#pragma unroll
        for (int p = 0; p < 2; ++p) {
            // layer 0: h1 = tanh(c1 + wlv*(lv+4)), one row per lane
            const float x = lv + 4.0f;
            const _Float16* c1p = &sm.c1[p][tid * CSTR];
#pragma unroll
            for (int j0 = 0; j0 < 64; j0 += 8) {
                v8h cv = *(const v8h*)(c1p + j0);
                v8h wv = *(const v8h*)(&sm.wlv[p][j0]);
                v8h o;
#pragma unroll
                for (int e = 0; e < 8; ++e)
                    o[e] = (_Float16)fast_tanh((float)cv[e] + (float)wv[e] * x);
                *(v8h*)(myrow + j0) = o;
            }
            // two 64x64 WMMA layers (in-wave, no barriers needed)
            hidden_layer(hb, sm.Wh[p][0], sm.bh[p][0], lane);
            hidden_layer(hb, sm.Wh[p][1], sm.bh[p][1], lane);
            // head: 64 -> 1 dot product, one row per lane
            float s = sm.b4[p];
#pragma unroll
            for (int j0 = 0; j0 < 64; j0 += 8) {
                v8h hv = *(const v8h*)(myrow + j0);
                v8h wv = *(const v8h*)(&sm.w4[p][j0]);
#pragma unroll
                for (int e = 0; e < 8; ++e) s += (float)hv[e] * (float)wv[e];
            }
            outm[p] = s;
        }
        const float drift_nn  = 0.2f * fast_tanh(outm[0]);
        const float diff_nn   = 1.5f * fast_sigmoid(outm[1]) + 0.1f;
        const float drift_phy = 0.1f * (-3.8f - lv);
        lv = lv + (drift_phy + drift_nn) * dt + diff_nn * dw;
        lv = fminf(fmaxf(lv, -7.0f), -0.5f);
        __builtin_nontemporal_store(__expf(lv), out + (size_t)step * BATCH + grow);
    }
}

extern "C" void kernel_launch(void* const* d_in, const int* in_sizes, int n_in,
                              void* d_out, int out_size, void* d_ws, size_t ws_size,
                              hipStream_t stream) {
    (void)in_sizes; (void)n_in; (void)d_ws; (void)ws_size; (void)out_size;
    Params prm;
    for (int p = 0; p < 2; ++p)
        for (int l = 0; l < 4; ++l) {
            prm.W[p][l]  = (const float*)d_in[4 + p * 8 + l * 2];
            prm.Bb[p][l] = (const float*)d_in[4 + p * 8 + l * 2 + 1];
        }
    nrs_kernel<<<BATCH / ROWS_PER_WG, 128, 0, stream>>>(
        (const float*)d_in[0], (const float*)d_in[1], (const float*)d_in[2],
        (const float*)d_in[3], prm, (float*)d_out);
}
